// AGCRN_80453327388934
// MI455X (gfx1250) — matrix operand order repaired and checked by reading, not compiled, not run on hardware
//
#include <hip/hip_runtime.h>

// =====================================================================
// AGCRN on MI455X (gfx1250, wave32, WMMA bf16 16x16x32, f32 accumulate)
//
//   S1 = softmax(relu(E E^T))           (2048x2048, bf16, L2-resident)
//   W[n] = E[n] . Wp  (per-node adaptive weights, bf16, materialized once)
//   per step t, per layer:
//     prep_xg   : xg[n][b][0:Cp]  = concat([x, (z*)h])   (k=0 / identity)
//     diffusion : xg[n][b][Cp:2Cp]= S1 @ (k=0 half)       (WMMA GEMM)
//     gconv     : out = xg_n @ W_n + bias (+sigmoid / +tanh+GRU update)
//
// LDS: B tiles stored TRANSPOSED ([col][k]) so both A and B fragments
// are two contiguous ds_load_b128 per lane (no scalar gather/packing).
// A tiles are filled with global_load_async_to_lds_b128 (ASYNCcnt).
// Workspace budget ~540 MB.
// =====================================================================

typedef __bf16 bf16;
typedef __bf16 v16bf __attribute__((ext_vector_type(16)));
typedef float  v8f   __attribute__((ext_vector_type(8)));

#define NN 2048   // nodes
#define BB 64     // batch
#define HH 64     // hidden
#define DDIM 16   // embedding dim
#define TT 12     // time steps / horizon

#define USE_ASYNC_LDS 1

// ---------------------------------------------------------------------
// CDNA5 async global->LDS copy (16B per lane). VDST (operand 0) is the
// wave-relative LDS address: flat shared pointers carry it in addr[31:0]
// (ISA 10.2: LDS aperture truncates to 32 bits). Tracked by ASYNCcnt.
// ---------------------------------------------------------------------
__device__ __forceinline__ void copy16_to_lds(void* lds_dst, const void* gsrc) {
#if USE_ASYNC_LDS
  unsigned l = (unsigned)(unsigned long long)lds_dst;
  unsigned long long g = (unsigned long long)gsrc;
  asm volatile("global_load_async_to_lds_b128 %0, %1, off"
               :: "v"(l), "v"(g) : "memory");
#else
  *reinterpret_cast<uint4*>(lds_dst) = *reinterpret_cast<const uint4*>(gsrc);
#endif
}
__device__ __forceinline__ void async_copy_wait() {
#if USE_ASYNC_LDS
  asm volatile("s_wait_asynccnt 0" ::: "memory");
#endif
}

// ---------------------------------------------------------------------
// Fragment loaders per CDNA5 ISA §7.12.2. Per-lane data is contiguous:
//  A (16x32, row-major tile, ld=ldm): lane m=l&15, kb=(l>>4)*8,
//     regs = row[kb..kb+7] ++ row[16+kb..16+kb+7].
//  B (32x16) from a TRANSPOSED tile BsT[col][k] (ld=ldk): lane n=l&15,
//     kb=(l>>4)*8, regs = BsT[n][kb..kb+7] ++ BsT[n][16+kb..16+kb+7].
// ---------------------------------------------------------------------
__device__ __forceinline__ v16bf frag_a_16x32(const bf16* tile, int ldm, int lane) {
  const int m  = lane & 15;
  const int kb = (lane >> 4) << 3;
  const bf16* p = tile + m * ldm + kb;
  v16bf a;
  reinterpret_cast<uint4*>(&a)[0] = *reinterpret_cast<const uint4*>(p);
  reinterpret_cast<uint4*>(&a)[1] = *reinterpret_cast<const uint4*>(p + 16);
  return a;
}

__device__ __forceinline__ v16bf frag_b_t32x16(const bf16* tileT, int ldk, int lane) {
  const int n  = lane & 15;
  const int kb = (lane >> 4) << 3;
  const bf16* p = tileT + n * ldk + kb;
  v16bf b;
  reinterpret_cast<uint4*>(&b)[0] = *reinterpret_cast<const uint4*>(p);
  reinterpret_cast<uint4*>(&b)[1] = *reinterpret_cast<const uint4*>(p + 16);
  return b;
}

// ---------------------------------------------------------------------
// S1 = softmax(relu(E E^T)) row-wise, store bf16. One block per row.
// ---------------------------------------------------------------------
__global__ __launch_bounds__(256) void supports_kernel(const float* __restrict__ E,
                                                       bf16* __restrict__ S1) {
  __shared__ float en[DDIM];
  __shared__ float row[NN];
  __shared__ float red[256];
  const int n = blockIdx.x, tid = threadIdx.x;
  if (tid < DDIM) en[tid] = E[n * DDIM + tid];
  __syncthreads();
  float lmax = -3.0e38f;
  for (int m = tid; m < NN; m += 256) {
    float d = 0.f;
#pragma unroll
    for (int k = 0; k < DDIM; ++k) d += en[k] * E[m * DDIM + k];
    d = fmaxf(d, 0.f);
    row[m] = d;
    lmax = fmaxf(lmax, d);
  }
  red[tid] = lmax;
  __syncthreads();
  for (int s = 128; s > 0; s >>= 1) {
    if (tid < s) red[tid] = fmaxf(red[tid], red[tid + s]);
    __syncthreads();
  }
  const float mx = red[0];
  __syncthreads();
  float lsum = 0.f;
  for (int m = tid; m < NN; m += 256) {
    float e = expf(row[m] - mx);
    row[m] = e;
    lsum += e;
  }
  red[tid] = lsum;
  __syncthreads();
  for (int s = 128; s > 0; s >>= 1) {
    if (tid < s) red[tid] += red[tid + s];
    __syncthreads();
  }
  const float inv = 1.f / red[0];
  for (int m = tid; m < NN; m += 256)
    S1[(size_t)n * NN + m] = (bf16)(row[m] * inv);
}

// ---------------------------------------------------------------------
// Per-node adaptive weights: Wn[n][kcp][o] = sum_d E[n,d] Wp[d,k,c,o]
// ---------------------------------------------------------------------
__global__ __launch_bounds__(256) void make_weights_kernel(const float* __restrict__ E,
    const float* __restrict__ Wp, bf16* __restrict__ Wn, int C, int Cp, int O, int KCp) {
  size_t idx = (size_t)blockIdx.x * 256 + threadIdx.x;
  size_t total = (size_t)NN * KCp * O;
  if (idx >= total) return;
  const int o   = (int)(idx % O);
  const int kcp = (int)((idx / O) % KCp);
  const int n   = (int)(idx / ((size_t)O * KCp));
  const int k = kcp / Cp, c = kcp % Cp;
  float s = 0.f;
  if (c < C) {
    const float* e = E + n * DDIM;
    const float* w = Wp + (((size_t)k * C + c) * O + o);
    const size_t dstride = (size_t)2 * C * O;
#pragma unroll
    for (int d = 0; d < DDIM; ++d) s += e[d] * w[(size_t)d * dstride];
  }
  Wn[idx] = (bf16)s;
}

__global__ __launch_bounds__(256) void make_bias_kernel(const float* __restrict__ E,
    const float* __restrict__ bp, float* __restrict__ bn, int O) {
  int idx = blockIdx.x * 256 + threadIdx.x;
  if (idx >= NN * O) return;
  const int o = idx % O, n = idx / O;
  float s = 0.f;
#pragma unroll
  for (int d = 0; d < DDIM; ++d) s += E[n * DDIM + d] * bp[d * O + o];
  bn[idx] = s;
}

// ---------------------------------------------------------------------
// Build the k=0 (identity) half of xg: concat([x, (z*)h]); zero pad +
// the (soon overwritten) k=1 half. xmode 0: x=source[b,t,n,0]; 1: x=h1.
// ---------------------------------------------------------------------
__global__ __launch_bounds__(256) void prep_xg_kernel(const float* __restrict__ xsrc,
    int xmode, int t, const float* __restrict__ h, const float* __restrict__ zr,
    bf16* __restrict__ xg, int Cx, int C, int Cp, int KCp) {
  size_t idx = (size_t)blockIdx.x * 256 + threadIdx.x;
  size_t total = (size_t)NN * BB * KCp;
  if (idx >= total) return;
  const int kcp = (int)(idx % KCp);
  const int b   = (int)((idx / KCp) % BB);
  const int n   = (int)(idx / ((size_t)KCp * BB));
  float val = 0.f;
  if (kcp < Cp) {
    const int c = kcp;
    if (c < C) {
      if (c < Cx) {
        val = (xmode == 0) ? xsrc[((size_t)b * TT + t) * NN + n]
                           : xsrc[((size_t)n * BB + b) * HH + c];
      } else {
        const int hh = c - Cx;
        float hv = h[((size_t)n * BB + b) * HH + hh];
        if (zr) hv *= zr[((size_t)n * BB + b) * 128 + hh];  // z gate
        val = hv;
      }
    }
  }
  xg[idx] = (bf16)val;
}

// ---------------------------------------------------------------------
// Diffusion GEMM: Y[n,(b,c)] = sum_m S1[n,m] * xg_k0[m,(b,c)]
// 128x128 macro-tile, K step 32, 8 waves x (2x4) WMMA tiles.
// ---------------------------------------------------------------------
template <int C, int Cp, int KCp>
__global__ __launch_bounds__(256) void diffusion_kernel(const bf16* __restrict__ S1,
                                                        bf16* __restrict__ xg) {
  __shared__ bf16 As[128][40];    // A row-major [row][k]
  __shared__ bf16 BsT[128][40];   // B transposed [col][k]
  const int tid  = threadIdx.x;
  const int lane = tid & 31, wave = tid >> 5;
  const int wm = wave & 3, wn = wave >> 2;
  const int n0   = blockIdx.y * 128;
  const int col0 = blockIdx.x * 128;
  constexpr int BC = BB * C;

  const v8f vz = {0.f, 0.f, 0.f, 0.f, 0.f, 0.f, 0.f, 0.f};
  v8f acc[2][4];
#pragma unroll
  for (int i = 0; i < 2; ++i)
#pragma unroll
    for (int j = 0; j < 4; ++j) acc[i][j] = vz;

  for (int kk = 0; kk < NN; kk += 32) {
    { // A tile: 128x32 bf16 from S1 via async global->LDS (2x16B/thread)
      const int row = tid >> 1, cs = (tid & 1) * 16;
      const bf16* g = S1 + (size_t)(n0 + row) * NN + kk + cs;
      copy16_to_lds(&As[row][cs], g);
      copy16_to_lds(&As[row][cs + 8], g + 8);
    }
    { // B tile: 32x128 gathered from xg k=0 half, stored transposed
      const int kr = tid >> 3;
      const int cb = (tid & 7) * 16;
      const int col = col0 + cb;
      const bf16* xrow = xg + ((size_t)(kk + kr) * BB) * KCp;
      if (col + 15 < BC) {
        int b = col / C, c = col % C;
#pragma unroll
        for (int j = 0; j < 16; ++j) {
          BsT[cb + j][kr] = xrow[(size_t)b * KCp + c];
          if (++c == C) { c = 0; ++b; }
        }
      } else {
        int b = 0, c = 0;
        if (col < BC) { b = col / C; c = col % C; }
#pragma unroll
        for (int j = 0; j < 16; ++j) {
          bf16 v = (bf16)0.f;
          if (col + j < BC) {
            v = xrow[(size_t)b * KCp + c];
            if (++c == C) { c = 0; ++b; }
          }
          BsT[cb + j][kr] = v;
        }
      }
    }
    async_copy_wait();
    __syncthreads();
    v16bf af[2], bfr[4];
#pragma unroll
    for (int i = 0; i < 2; ++i) af[i] = frag_a_16x32(&As[wm * 32 + i * 16][0], 40, lane);
#pragma unroll
    for (int j = 0; j < 4; ++j) bfr[j] = frag_b_t32x16(&BsT[wn * 64 + j * 16][0], 40, lane);
#pragma unroll
    for (int i = 0; i < 2; ++i)
#pragma unroll
      for (int j = 0; j < 4; ++j)
        acc[i][j] = __builtin_amdgcn_wmma_f32_16x16x32_bf16(
            false, af[i], false, bfr[j], (short)0, acc[i][j], false, false);
    __syncthreads();
  }

  // Epilogue: D layout lane=(n, half), VGPR v = row mb+v. Write k=1 half.
  const int mb   = (lane >> 4) * 8;
  const int ncol = lane & 15;
#pragma unroll
  for (int i = 0; i < 2; ++i) {
#pragma unroll
    for (int j = 0; j < 4; ++j) {
      const int col_g = col0 + wn * 64 + j * 16 + ncol;
      if (col_g >= BC) continue;
      const int b = col_g / C, c = col_g % C;
#pragma unroll
      for (int v = 0; v < 8; ++v) {
        const int n_g = n0 + wm * 32 + i * 16 + mb + v;
        xg[((size_t)n_g * BB + b) * KCp + Cp + c] = (bf16)acc[i][j][v];
      }
    }
  }
}

// ---------------------------------------------------------------------
// Per-node gconv GEMM: out[b,o] = xg_n[b,:] . W_n[:,o] + bias_n[o]
// MODE 0: zr = sigmoid(out)                  (gate, O=128)
// MODE 1: hc = tanh(out); h = r*h + (1-r)*hc (candidate+update, O=64)
// ---------------------------------------------------------------------
template <int KCp, int O, int MODE>
__global__ __launch_bounds__(256) void gconv_kernel(const bf16* __restrict__ xg,
    const bf16* __restrict__ W, const float* __restrict__ bias,
    float* __restrict__ zr, float* __restrict__ h) {
  __shared__ bf16 As[64][40];     // A row-major [b][k]
  __shared__ bf16 BsT[128][40];   // weights transposed [o][k]
  const int tid  = threadIdx.x;
  const int lane = tid & 31, wave = tid >> 5;
  const int wm  = wave & 3;
  const int wn0 = (wave >> 2) * (O >> 1);
  constexpr int NTILES = O >> 5;  // 4 (O=128) or 2 (O=64)
  const int n = blockIdx.x;

  const v8f vz = {0.f, 0.f, 0.f, 0.f, 0.f, 0.f, 0.f, 0.f};
  v8f acc[NTILES];
#pragma unroll
  for (int j = 0; j < NTILES; ++j) acc[j] = vz;

  for (int kk = 0; kk < KCp; kk += 32) {
    { // A: 64x32 from xg[n] via async global->LDS (16B/thread)
      const int row = tid >> 2, cs = (tid & 3) * 8;
      copy16_to_lds(&As[row][cs], xg + ((size_t)n * BB + row) * KCp + kk + cs);
    }
    { // B: 32xO rows of W[n], stored transposed into BsT
      constexpr int PER = O >> 3;  // 16B chunks per k-row
#pragma unroll
      for (int idx = tid; idx < 4 * O; idx += 256) {
        const int kr = idx / PER, os = (idx % PER) * 8;
        uint4 w4 = *reinterpret_cast<const uint4*>(W + ((size_t)n * KCp + kk + kr) * O + os);
        const bf16* wv = reinterpret_cast<const bf16*>(&w4);
#pragma unroll
        for (int i = 0; i < 8; ++i) BsT[os + i][kr] = wv[i];
      }
    }
    async_copy_wait();
    __syncthreads();
    v16bf a = frag_a_16x32(&As[wm * 16][0], 40, lane);
#pragma unroll
    for (int j = 0; j < NTILES; ++j) {
      v16bf b = frag_b_t32x16(&BsT[wn0 + j * 16][0], 40, lane);
      acc[j] = __builtin_amdgcn_wmma_f32_16x16x32_bf16(
          false, a, false, b, (short)0, acc[j], false, false);
    }
    __syncthreads();
  }

  const int mb   = (lane >> 4) * 8;
  const int ncol = lane & 15;
#pragma unroll
  for (int j = 0; j < NTILES; ++j) {
    const int o = wn0 + j * 16 + ncol;
    const float bv = bias[(size_t)n * O + o];
#pragma unroll
    for (int v = 0; v < 8; ++v) {
      const int b = wm * 16 + mb + v;
      const float val = acc[j][v] + bv;
      const size_t base = (size_t)n * BB + b;
      if (MODE == 0) {
        zr[base * 128 + o] = 1.f / (1.f + expf(-val));
      } else {
        const float hc   = tanhf(val);
        const float r    = zr[base * 128 + 64 + o];
        const float hold = h[base * HH + o];
        h[base * HH + o] = r * hold + (1.f - r) * hc;
      }
    }
  }
}

// ---------------------------------------------------------------------
// Output: out[b, c, n] = h2[n][b][:] . conv_w[c][:] + conv_b[c]
// ---------------------------------------------------------------------
__global__ __launch_bounds__(256) void output_kernel(const float* __restrict__ h2,
    const float* __restrict__ w, const float* __restrict__ cb, float* __restrict__ out) {
  int idx = blockIdx.x * 256 + threadIdx.x;
  if (idx >= BB * TT * NN) return;
  const int n = idx % NN;
  const int c = (idx / NN) % TT;
  const int b = idx / (NN * TT);
  const float* hp = h2 + ((size_t)n * BB + b) * HH;
  const float* wp = w + c * HH;
  float s = cb[c];
#pragma unroll
  for (int k = 0; k < HH; ++k) s += hp[k] * wp[k];
  out[idx] = s;
}

// =====================================================================
extern "C" void kernel_launch(void* const* d_in, const int* in_sizes, int n_in,
                              void* d_out, int out_size, void* d_ws, size_t ws_size,
                              hipStream_t stream) {
  const float* src   = (const float*)d_in[0];
  const float* E     = (const float*)d_in[1];
  const float* Wg0p  = (const float*)d_in[2];
  const float* bg0p  = (const float*)d_in[3];
  const float* Wu0p  = (const float*)d_in[4];
  const float* bu0p  = (const float*)d_in[5];
  const float* Wg1p  = (const float*)d_in[6];
  const float* bg1p  = (const float*)d_in[7];
  const float* Wu1p  = (const float*)d_in[8];
  const float* bu1p  = (const float*)d_in[9];
  const float* convw = (const float*)d_in[10];
  const float* convb = (const float*)d_in[11];
  float* out = (float*)d_out;

  // Workspace carve-out (~540 MB total, 256B aligned slices)
  char* p = (char*)d_ws;
  auto carve = [&](size_t bytes) -> void* {
    void* r = (void*)p;
    p += (bytes + 255) & ~(size_t)255;
    return r;
  };
  bf16*  S1  = (bf16*)carve((size_t)NN * NN * 2);            //   8 MB
  bf16*  W0g = (bf16*)carve((size_t)NN * 160 * 128 * 2);     //  84 MB
  bf16*  W0u = (bf16*)carve((size_t)NN * 160 * 64 * 2);      //  42 MB
  bf16*  W1g = (bf16*)carve((size_t)NN * 256 * 128 * 2);     // 134 MB
  bf16*  W1u = (bf16*)carve((size_t)NN * 256 * 64 * 2);      //  67 MB
  float* bg0 = (float*)carve((size_t)NN * 128 * 4);
  float* bu0 = (float*)carve((size_t)NN * 64 * 4);
  float* bg1 = (float*)carve((size_t)NN * 128 * 4);
  float* bu1 = (float*)carve((size_t)NN * 64 * 4);
  bf16*  xg  = (bf16*)carve((size_t)NN * BB * 256 * 2);      //  67 MB
  float* zr  = (float*)carve((size_t)NN * BB * 128 * 4);     //  67 MB
  float* h1  = (float*)carve((size_t)NN * BB * HH * 4);      //  33 MB
  float* h2  = (float*)carve((size_t)NN * BB * HH * 4);      //  33 MB

  hipMemsetAsync(h1, 0, (size_t)NN * BB * HH * 4, stream);
  hipMemsetAsync(h2, 0, (size_t)NN * BB * HH * 4, stream);

  auto blocks = [](size_t total) { return (unsigned)((total + 255) / 256); };

  supports_kernel<<<NN, 256, 0, stream>>>(E, S1);

  make_weights_kernel<<<blocks((size_t)NN * 160 * 128), 256, 0, stream>>>(E, Wg0p, W0g, 65, 80, 128, 160);
  make_weights_kernel<<<blocks((size_t)NN * 160 * 64),  256, 0, stream>>>(E, Wu0p, W0u, 65, 80, 64, 160);
  make_weights_kernel<<<blocks((size_t)NN * 256 * 128), 256, 0, stream>>>(E, Wg1p, W1g, 128, 128, 128, 256);
  make_weights_kernel<<<blocks((size_t)NN * 256 * 64),  256, 0, stream>>>(E, Wu1p, W1u, 128, 128, 64, 256);
  make_bias_kernel<<<blocks((size_t)NN * 128), 256, 0, stream>>>(E, bg0p, bg0, 128);
  make_bias_kernel<<<blocks((size_t)NN * 64),  256, 0, stream>>>(E, bu0p, bu0, 64);
  make_bias_kernel<<<blocks((size_t)NN * 128), 256, 0, stream>>>(E, bg1p, bg1, 128);
  make_bias_kernel<<<blocks((size_t)NN * 64),  256, 0, stream>>>(E, bu1p, bu1, 64);

  for (int t = 0; t < TT; ++t) {
    // ---- layer 0: x = source[:, t], state h1, C=65 (Cp=80, KCp=160) ----
    prep_xg_kernel<<<blocks((size_t)NN * BB * 160), 256, 0, stream>>>(src, 0, t, h1, nullptr, xg, 1, 65, 80, 160);
    diffusion_kernel<65, 80, 160><<<dim3(33, 16), 256, 0, stream>>>(S1, xg);
    gconv_kernel<160, 128, 0><<<NN, 256, 0, stream>>>(xg, W0g, bg0, zr, h1);

    prep_xg_kernel<<<blocks((size_t)NN * BB * 160), 256, 0, stream>>>(src, 0, t, h1, zr, xg, 1, 65, 80, 160);
    diffusion_kernel<65, 80, 160><<<dim3(33, 16), 256, 0, stream>>>(S1, xg);
    gconv_kernel<160, 64, 1><<<NN, 256, 0, stream>>>(xg, W0u, bu0, zr, h1);

    // ---- layer 1: x = h1 (updated), state h2, C=128 (KCp=256) ----
    prep_xg_kernel<<<blocks((size_t)NN * BB * 256), 256, 0, stream>>>(h1, 1, t, h2, nullptr, xg, 64, 128, 128, 256);
    diffusion_kernel<128, 128, 256><<<dim3(64, 16), 256, 0, stream>>>(S1, xg);
    gconv_kernel<256, 128, 0><<<NN, 256, 0, stream>>>(xg, W1g, bg1, zr, h2);

    prep_xg_kernel<<<blocks((size_t)NN * BB * 256), 256, 0, stream>>>(h1, 1, t, h2, zr, xg, 64, 128, 128, 256);
    diffusion_kernel<128, 128, 256><<<dim3(64, 16), 256, 0, stream>>>(S1, xg);
    gconv_kernel<256, 64, 1><<<NN, 256, 0, stream>>>(xg, W1u, bu1, zr, h2);
  }

  output_kernel<<<blocks((size_t)BB * TT * NN), 256, 0, stream>>>(h2, convw, convb, out);
}
